// LyraGemma3Attention_6605659701808
// MI455X (gfx1250) — matrix operand order repaired and checked by reading, not compiled, not run on hardware
//
#include <hip/hip_runtime.h>
#include <hip/hip_bf16.h>
#include <math.h>

typedef __attribute__((ext_vector_type(16))) __bf16 v16bf;
typedef __attribute__((ext_vector_type(8)))  float  v8f;

namespace {
constexpr int   kB    = 2;
constexpr int   kS    = 2048;
constexpr int   kD    = 2560;
constexpr int   kH    = 8;
constexpr int   kKVH  = 4;
constexpr int   kHD   = 256;
constexpr int   kNRep = kH / kKVH;
constexpr int   kWin  = 512;
constexpr float kScale = 0.0625f;      // 256^-0.5
constexpr float kCap   = 50.0f;
constexpr float kNeg   = -1.0e9f;
constexpr float kEps   = 1e-6f;
}

// ---------------------------------------------------------------------------
// Zero-fill (probs tensor must be 0 outside the attention window).
// ---------------------------------------------------------------------------
__global__ __launch_bounds__(256) void zero_f32(float* __restrict__ p, size_t n) {
  size_t i = (size_t)blockIdx.x * blockDim.x + threadIdx.x;
  const size_t stride = (size_t)gridDim.x * blockDim.x;
  for (; i < n; i += stride) p[i] = 0.0f;
}

// ---------------------------------------------------------------------------
// Generic C[M,N] = A[M,K] * W[N,K]^T, f32 in/out, bf16 WMMA accumulate-f32.
// Block: 256 threads = 8 waves (2x4 wave grid) -> 32x64 output tile.
// K-step 64: float4 global loads, 2 chained WMMAs per barrier pair.
// ---------------------------------------------------------------------------
__global__ __launch_bounds__(256) void gemm_atb_bf16(
    const float* __restrict__ A, const float* __restrict__ W, float* __restrict__ C,
    int M, int N, int K)
{
  constexpr int TK = 64;
  __shared__ __bf16 As[32][TK + 8];
  __shared__ __bf16 Ws[64][TK + 8];
  const int tid  = threadIdx.x;
  const int wave = tid >> 5, lane = tid & 31;
  const int half = lane >> 4, l = lane & 15;
  const int m0 = blockIdx.y * 32;
  const int n0 = blockIdx.x * 64;
  const int wm = (wave >> 2) * 16;
  const int wn = (wave & 3) * 16;
  v8f acc = {};
  for (int k0 = 0; k0 < K; k0 += TK) {
    // A tile: 32 x 64 f32 = 512 float4 (16B, aligned: K % 4 == 0)
#pragma unroll
    for (int i = 0; i < 2; ++i) {
      const int idx = tid + i * 256;
      const int r = idx >> 4, c4 = (idx & 15) * 4;
      const float4 t = *reinterpret_cast<const float4*>(
          A + (size_t)(m0 + r) * K + (k0 + c4));
      __bf16* dp = &As[r][c4];
      dp[0] = (__bf16)t.x; dp[1] = (__bf16)t.y;
      dp[2] = (__bf16)t.z; dp[3] = (__bf16)t.w;
    }
    // W tile: 64 x 64 f32 = 1024 float4
#pragma unroll
    for (int i = 0; i < 4; ++i) {
      const int idx = tid + i * 256;
      const int r = idx >> 4, c4 = (idx & 15) * 4;
      const float4 t = *reinterpret_cast<const float4*>(
          W + (size_t)(n0 + r) * K + (k0 + c4));
      __bf16* dp = &Ws[r][c4];
      dp[0] = (__bf16)t.x; dp[1] = (__bf16)t.y;
      dp[2] = (__bf16)t.z; dp[3] = (__bf16)t.w;
    }
    if (k0 + TK < K)                            // gfx1250 global_prefetch_b8
      __builtin_prefetch(A + (size_t)(m0 + lane) * K + (k0 + TK), 0, 1);
    __syncthreads();

#pragma unroll
    for (int d0 = 0; d0 < TK; d0 += 32) {
      v16bf af = {}, bw = {};
      // A 16x32 frag: lane = row; lanes<16 K 0..15, lanes>=16 K 16..31
#pragma unroll
      for (int j = 0; j < 16; ++j)
        af[j] = As[wm + l][d0 + half * 16 + j];
      // B 32x16 frag: lane = column; lanes<16 K 0..15, lanes>=16 K 16..31
#pragma unroll
      for (int j = 0; j < 16; ++j)
        bw[j] = Ws[wn + l][d0 + half * 16 + j];
      acc = __builtin_amdgcn_wmma_f32_16x16x32_bf16(false, af, false, bw,
                                                    (short)0, acc, false, false);
    }
    __syncthreads();
  }
  // C/D layout: VGPR r -> M = r + 8*half, N = lane&15
#pragma unroll
  for (int r = 0; r < 8; ++r)
    C[(size_t)(m0 + wm + r + half * 8) * N + (n0 + wn + l)] = acc[r];
}

// ---------------------------------------------------------------------------
// Fused RMS-norm (over HD=256) + RoPE. One block (128 threads) per (b,s,head)
// row; thread t owns elements t and t+128 (the rotate_half pair).
// ---------------------------------------------------------------------------
__global__ __launch_bounds__(128) void rmsnorm_rope(
    float* __restrict__ x, const float* __restrict__ cs, const float* __restrict__ sn,
    const float* __restrict__ w, int nheads)
{
  const int row = blockIdx.x;            // (b*S + s)*nheads + h
  const int bs  = row / nheads;          // b*S + s
  const int t   = threadIdx.x;
  float* p = x + (size_t)row * kHD;
  const float x1 = p[t], x2 = p[t + 128];
  __shared__ float red[128];
  red[t] = x1 * x1 + x2 * x2;
  __syncthreads();
  for (int o = 64; o > 0; o >>= 1) {
    if (t < o) red[t] += red[t + o];
    __syncthreads();
  }
  const float inv = rsqrtf(red[0] * (1.0f / kHD) + kEps);
  const float n1 = x1 * inv * (1.0f + w[t]);
  const float n2 = x2 * inv * (1.0f + w[t + 128]);
  const size_t cb = (size_t)bs * kHD;
  p[t]       = n1 * cs[cb + t]       - n2 * sn[cb + t];
  p[t + 128] = n2 * cs[cb + t + 128] + n1 * sn[cb + t + 128];
}

// ---------------------------------------------------------------------------
// Windowed attention with softcap + GQA. One block per (b, h, 16-query tile).
// 8 waves. Phase1: QK^T via bf16 WMMA (Q staged in LDS, K read b128-contig).
// Phase2: row softmax + probs store. Phase3: PV with V chunks staged
// coalesced into LDS (overlaying the Q buffer), 2 WMMAs per chunk per wave.
// ---------------------------------------------------------------------------
__global__ __launch_bounds__(256) void attn_kernel(
    const float* __restrict__ q, const float* __restrict__ k, const float* __restrict__ v,
    float* __restrict__ probs, float* __restrict__ ctx)
{
  constexpr int MAXC = 34 * 16;            // 33 key tiles + 1 pad tile
  constexpr int QVS  = kHD + 12;           // 268 bf16 = 134 dw: 134%64=6, 16*134%64=32
  __shared__ __align__(16) __bf16 QV[32 * QVS];   // phase1: Q(16 rows); phase3: V(32 rows)
  __shared__ float Sc[16][MAXC];
  const int tid  = threadIdx.x;
  const int wave = tid >> 5, lane = tid & 31;
  const int half = lane >> 4, l = lane & 15;
  const int qt = blockIdx.x, h = blockIdx.y, b = blockIdx.z;
  const int kh = h / kNRep;
  const int i0 = qt * 16;
  const int kt0  = (qt > kWin / 16) ? qt - kWin / 16 : 0;
  const int n_kt = qt - kt0 + 1;           // <= 33 key tiles
  const int nkt2 = (n_kt + 1) & ~1;        // padded to even for K=32 chunks

  // Stage Q tile (16 x 256) into LDS as bf16, float4-coalesced.
#pragma unroll
  for (int i = 0; i < 4; ++i) {
    const int idx = tid + i * 256;         // 0..1023 float4
    const int r = idx >> 6, c4 = (idx & 63) * 4;
    const float4 t = *reinterpret_cast<const float4*>(
        q + (((size_t)b * kS + (i0 + r)) * kH + h) * kHD + c4);
    __bf16* dp = &QV[r * QVS + c4];
    dp[0] = (__bf16)t.x; dp[1] = (__bf16)t.y;
    dp[2] = (__bf16)t.z; dp[3] = (__bf16)t.w;
  }
  if (nkt2 != n_kt)                        // zero the pad key tile
    Sc[tid >> 4][n_kt * 16 + (tid & 15)] = 0.0f;
  __syncthreads();

  // --- Phase 1: scores = softcap(scale * Q K^T) + window mask -------------
  for (int kti = wave; kti < n_kt; kti += 8) {
    const int jb = (kt0 + kti) * 16;
    v8f acc = {};
#pragma unroll
    for (int d0 = 0; d0 < kHD; d0 += 32) {
      v16bf af = {}, bw = {};
#pragma unroll
      for (int j = 0; j < 16; ++j)
        af[j] = QV[l * QVS + d0 + half * 16 + j];
      // B = K^T: lane = key column (row of K), 16 contiguous d-values (4x b128).
      const float* krow =
          k + (((size_t)b * kS + (jb + l)) * kKVH + kh) * kHD + d0 + half * 16;
#pragma unroll
      for (int e = 0; e < 16; ++e) bw[e] = (__bf16)krow[e];
      acc = __builtin_amdgcn_wmma_f32_16x16x32_bf16(false, af, false, bw,
                                                    (short)0, acc, false, false);
    }
#pragma unroll
    for (int r = 0; r < 8; ++r) {
      const int qi = i0 + r + half * 8;
      const int kj = jb + l;
      float s = acc[r] * kScale;
      s = kCap * tanhf(s * (1.0f / kCap));
      const bool ok = (kj <= qi) && (qi - kj < kWin);
      Sc[r + half * 8][kti * 16 + l] = ok ? s : kNeg;
    }
  }
  __syncthreads();

  // --- Phase 2: row softmax, write probs ----------------------------------
  const int ncol = n_kt * 16;
  for (int m = wave * 2; m < wave * 2 + 2; ++m) {
    const int qi = i0 + m;
    float mx = kNeg;
    for (int c = lane; c < ncol; c += 32) mx = fmaxf(mx, Sc[m][c]);
#pragma unroll
    for (int o = 16; o > 0; o >>= 1) mx = fmaxf(mx, __shfl_xor(mx, o, 32));
    float sum = 0.0f;
    for (int c = lane; c < ncol; c += 32) {
      const float e = __expf(Sc[m][c] - mx);
      Sc[m][c] = e;
      sum += e;
    }
#pragma unroll
    for (int o = 16; o > 0; o >>= 1) sum += __shfl_xor(sum, o, 32);
    const float inv = 1.0f / sum;
    float* prow = probs + (((size_t)b * kH + h) * kS + qi) * kS + kt0 * 16;
    for (int c = lane; c < ncol; c += 32) {
      const float pv = Sc[m][c] * inv;
      Sc[m][c] = pv;
      prow[c]  = pv;
    }
  }

  // --- Phase 3: ctx = P V (keys in chunks of 32, V staged in LDS) ---------
  const int n_jc = nkt2 / 2;
  const int ct0 = wave, ct1 = wave + 8;    // each wave owns 2 HD column tiles
  v8f acc0 = {}, acc1 = {};
  for (int jc = 0; jc < n_jc; ++jc) {
    __syncthreads();                       // previous chunk fully consumed
    // Stage V chunk (32 keys x 256 d) -> LDS bf16, float4-coalesced.
#pragma unroll
    for (int i = 0; i < 8; ++i) {
      const int idx = tid + i * 256;       // 0..2047 float4
      const int r = idx >> 6, c4 = (idx & 63) * 4;
      int jg = kt0 * 16 + jc * 32 + r;
      jg = (jg < kS) ? jg : (kS - 1);      // pad region: P == 0 there
      const float4 t = *reinterpret_cast<const float4*>(
          v + (((size_t)b * kS + jg) * kKVH + kh) * kHD + c4);
      __bf16* dp = &QV[r * QVS + c4];
      dp[0] = (__bf16)t.x; dp[1] = (__bf16)t.y;
      dp[2] = (__bf16)t.z; dp[3] = (__bf16)t.w;
    }
    __syncthreads();
    const int cb = jc * 32;
    v16bf af = {};
#pragma unroll
    for (int j = 0; j < 16; ++j)
      af[j] = (__bf16)Sc[l][cb + half * 16 + j];
    v16bf bw0 = {}, bw1 = {};
#pragma unroll
    for (int e = 0; e < 16; ++e) {
      bw0[e] = QV[(half * 16 + e) * QVS + ct0 * 16 + l];
      bw1[e] = QV[(half * 16 + e) * QVS + ct1 * 16 + l];
    }
    acc0 = __builtin_amdgcn_wmma_f32_16x16x32_bf16(false, af, false, bw0,
                                                   (short)0, acc0, false, false);
    acc1 = __builtin_amdgcn_wmma_f32_16x16x32_bf16(false, af, false, bw1,
                                                   (short)0, acc1, false, false);
  }
#pragma unroll
  for (int r = 0; r < 8; ++r) {
    const int qi = i0 + r + half * 8;
    const size_t base = (((size_t)b * kS + qi) * kH + h) * kHD;
    ctx[base + ct0 * 16 + l] = acc0[r];
    ctx[base + ct1 * 16 + l] = acc1[r];
  }
}

// ---------------------------------------------------------------------------
extern "C" void kernel_launch(void* const* d_in, const int* in_sizes, int n_in,
                              void* d_out, int out_size, void* d_ws, size_t ws_size,
                              hipStream_t stream) {
  (void)in_sizes; (void)n_in; (void)out_size; (void)ws_size;
  const float* hidden = (const float*)d_in[0];
  const float* cosb   = (const float*)d_in[1];
  const float* sinb   = (const float*)d_in[2];
  // d_in[3] attention_mask: windowed-causal mask recomputed analytically in-kernel
  const float* q_w = (const float*)d_in[4];
  const float* k_w = (const float*)d_in[5];
  const float* v_w = (const float*)d_in[6];
  const float* o_w = (const float*)d_in[7];
  const float* qn  = (const float*)d_in[8];
  const float* kn  = (const float*)d_in[9];

  float* q   = (float*)d_ws;                                   // (b,s,h,hd)
  float* kk  = q  + (size_t)kB * kS * kH * kHD;                // (b,s,kvh,hd)
  float* vv  = kk + (size_t)kB * kS * kKVH * kHD;
  float* ctx = vv + (size_t)kB * kS * kKVH * kHD;              // (b,s,h,hd)

  float* out   = (float*)d_out;                                // (b,s,D)
  float* probs = out + (size_t)kB * kS * kD;                   // (b,h,s,s)

  zero_f32<<<2048, 256, 0, stream>>>(probs, (size_t)kB * kH * kS * kS);

  const int M = kB * kS;                                       // 4096 rows
  gemm_atb_bf16<<<dim3((kH  * kHD) / 64, M / 32), 256, 0, stream>>>(hidden, q_w, q,  M, kH  * kHD, kD);
  gemm_atb_bf16<<<dim3((kKVH * kHD) / 64, M / 32), 256, 0, stream>>>(hidden, k_w, kk, M, kKVH * kHD, kD);
  gemm_atb_bf16<<<dim3((kKVH * kHD) / 64, M / 32), 256, 0, stream>>>(hidden, v_w, vv, M, kKVH * kHD, kD);

  rmsnorm_rope<<<M * kH,   128, 0, stream>>>(q,  cosb, sinb, qn, kH);
  rmsnorm_rope<<<M * kKVH, 128, 0, stream>>>(kk, cosb, sinb, kn, kKVH);

  attn_kernel<<<dim3(kS / 16, kH, kB), 256, 0, stream>>>(q, kk, vv, probs, ctx);

  gemm_atb_bf16<<<dim3(kD / 64, M / 32), 256, 0, stream>>>(ctx, o_w, out, M, kD, kH * kHD);
}
